// ResidualVectorQuantizer_20160576487974
// MI455X (gfx1250) — matrix operand order repaired and checked by reading, not compiled, not run on hardware
//
#include <hip/hip_runtime.h>
#include <math.h>

typedef _Float16 f16;
typedef _Float16 v16h __attribute__((ext_vector_type(16)));
typedef _Float16 v8h  __attribute__((ext_vector_type(8)));
typedef float    v8f  __attribute__((ext_vector_type(8)));
typedef int      v4i  __attribute__((ext_vector_type(4)));

#define BT_   131072
#define DIM_  96
#define KCB_  384
#define NCB_  6
#define PROJ_ 192
#define W2V_  768

// workspace byte offsets (all 256B aligned)
#define WS_RESID 0u                       // BT*DIM f32         = 50,331,648 B
#define WS_CBH   50331648u                // NCB*KCB*DIM f16    =    442,368 B
#define WS_W1T   50774016u                // PROJ*DIM f16       =     36,864 B
#define WS_W2T   50810880u                // W2V*PROJ f16       =    294,912 B
#define WS_NORM  51105792u                // NCB*KCB f32        =      9,216 B
#define WS_ACC   51115008u                // 2 f32 accumulators

// ---- CDNA5 async global->LDS copy (guarded; falls back to sync b128) ------
#if defined(__has_builtin)
#if __has_builtin(__builtin_amdgcn_global_load_async_to_lds_b128)
#define HAVE_ASYNC_LDS 1
#endif
#if __has_builtin(__builtin_amdgcn_s_wait_asynccnt)
#define HAVE_WAIT_ASYNC 1
#endif
#endif

typedef __attribute__((address_space(1))) v4i as1_v4i;   // global (prints __device__)
typedef __attribute__((address_space(3))) v4i as3_v4i;   // LDS

__device__ inline void async_cp16(const void* g, void* l) {
#ifdef HAVE_ASYNC_LDS
  __builtin_amdgcn_global_load_async_to_lds_b128((as1_v4i*)g, (as3_v4i*)l, 0, 0);
#else
  *(v4i*)l = *(const v4i*)g;
#endif
}

__device__ inline void wait_async() {
#ifdef HAVE_ASYNC_LDS
#ifdef HAVE_WAIT_ASYNC
  __builtin_amdgcn_s_wait_asynccnt(0);
#else
  asm volatile("s_wait_asynccnt 0x0" ::: "memory");
#endif
#endif
}

// ---- WMMA fragment loaders (CDNA5 16x16x32 f16 layouts) -------------------
// A (16xK): lane holds row m=lane&15; a[0..15]=K{koff..koff+15},
// koff = 0 (lanes 0-15) or 16 (lanes 16-31).
__device__ inline v16h load_a16(const f16* __restrict__ p, int lane) {
  const int koff = (lane < 16) ? 0 : 16;
  v8h lo = *(const v8h*)(p + koff);
  v8h hi = *(const v8h*)(p + koff + 8);
  v16h a;
#pragma unroll
  for (int j = 0; j < 8; ++j) { a[j] = lo[j]; a[8 + j] = hi[j]; }
  return a;
}
// B (Kx16) from f16: 16 contiguous K halves at boff = 0/16 per lane half.
__device__ inline v16h load_b16(const f16* __restrict__ p, int lane) {
  const int boff = (lane < 16) ? 0 : 16;
  return *(const v16h*)(p + boff);
}
// B (Kx16) built from f32 LDS data with on-the-fly f16 convert.
__device__ inline v16h load_b16_f32(const float* p, int lane) {
  const int boff = (lane < 16) ? 0 : 16;
  v16h b;
#pragma unroll
  for (int j = 0; j < 16; ++j) b[j] = (f16)p[boff + j];
  return b;
}

// ---- prep kernels ----------------------------------------------------------
__global__ void init_kernel(const float* __restrict__ z, float* __restrict__ resid,
                            float* __restrict__ acc) {
  size_t n = (size_t)BT_ * DIM_;
  for (size_t i = (size_t)blockIdx.x * blockDim.x + threadIdx.x; i < n;
       i += (size_t)gridDim.x * blockDim.x)
    resid[i] = z[i];
  if (blockIdx.x == 0 && threadIdx.x == 0) { acc[0] = 0.0f; acc[1] = 0.0f; }
}

__global__ void cvt_cb_kernel(const float* __restrict__ cb, f16* __restrict__ cbh) {
  size_t i = (size_t)blockIdx.x * blockDim.x + threadIdx.x;
  if (i < (size_t)NCB_ * KCB_ * DIM_) cbh[i] = (f16)cb[i];
}

__global__ void norms_kernel(const float* __restrict__ cb, float* __restrict__ norms) {
  int c = blockIdx.x * blockDim.x + threadIdx.x;
  if (c >= NCB_ * KCB_) return;
  const float* p = cb + (size_t)c * DIM_;
  float s = 0.0f;
#pragma unroll 4
  for (int d = 0; d < DIM_; ++d) s += p[d] * p[d];
  norms[c] = s;
}

__global__ void cvt_w_kernel(const float* __restrict__ w1, const float* __restrict__ w2,
                             f16* __restrict__ w1t, f16* __restrict__ w2t) {
  int i = blockIdx.x * blockDim.x + threadIdx.x;
  const int n1 = PROJ_ * DIM_;
  const int n2 = W2V_ * PROJ_;
  if (i < n1) {
    int n = i / DIM_, k = i % DIM_;
    w1t[i] = (f16)w1[(size_t)k * PROJ_ + n];
  } else if (i < n1 + n2) {
    int j = i - n1;
    int n = j / PROJ_, k = j % PROJ_;
    w2t[j] = (f16)w2[(size_t)k * W2V_ + n];
  }
}

// ---- one RVQ stage ---------------------------------------------------------
// Async-copies the f32 residual tile to LDS (single global read), builds f16 B
// fragments in-register, streams 24 codebook A tiles through WMMA, argmin per
// lane (token column), then residual update + loss from the LDS copy.
// Stage 5 additionally emits z_q_total = z - residual_final.
__global__ __launch_bounds__(256) void vq_stage_kernel(
    const float* __restrict__ cb, const f16* __restrict__ cbh,
    const float* __restrict__ norms, const float* __restrict__ z,
    float* __restrict__ resid, float* __restrict__ zqt,
    int* __restrict__ codes, float* __restrict__ acc, int stage) {
  __shared__ alignas(16) float sRf[8 * 16 * DIM_];  // per-wave f32 residual tiles
  __shared__ float sN[KCB_];

  const int tid = threadIdx.x;
  const int wave = tid >> 5;
  const int lane = tid & 31;
  const int tok0 = (blockIdx.x * 8 + wave) * 16;
  float* myR = sRf + wave * (16 * DIM_);
  const float* rsrc = resid + (size_t)tok0 * DIM_;

  // async residual tile -> LDS (384 x 16B chunks per wave)
  for (int c = lane; c < (16 * DIM_) / 4; c += 32)
    async_cp16(rsrc + 4 * c, myR + 4 * c);
  for (int i = tid; i < KCB_; i += 256) sN[i] = norms[stage * KCB_ + i];
  wait_async();
  __syncthreads();

  const int mrow = lane & 15;
  const int chalf = (lane < 16) ? 0 : 8;
  const f16* cbs = cbh + (size_t)stage * KCB_ * DIM_;

  // build the three K-step B fragments once (f32 LDS -> f16 regs)
  v16h bf0 = load_b16_f32(myR + mrow * DIM_ + 0, lane);
  v16h bf1 = load_b16_f32(myR + mrow * DIM_ + 32, lane);
  v16h bf2 = load_b16_f32(myR + mrow * DIM_ + 64, lane);

  float minv = 3.0e38f;
  int mini = 0;
  for (int t = 0; t < 24; ++t) {
    const f16* arow = cbs + (size_t)(t * 16 + mrow) * DIM_;
    __builtin_prefetch(arow + 16 * DIM_, 0, 0);
    v8f dacc = {};
    dacc = __builtin_amdgcn_wmma_f32_16x16x32_f16(false, load_a16(arow, lane),
                                                  false, bf0, (short)0, dacc, false, false);
    dacc = __builtin_amdgcn_wmma_f32_16x16x32_f16(false, load_a16(arow + 32, lane),
                                                  false, bf1, (short)0, dacc, false, false);
    dacc = __builtin_amdgcn_wmma_f32_16x16x32_f16(false, load_a16(arow + 64, lane),
                                                  false, bf2, (short)0, dacc, false, false);
#pragma unroll
    for (int v = 0; v < 8; ++v) {
      int code = t * 16 + v + chalf;
      float d = sN[code] - 2.0f * dacc[v];  // ||r||^2 constant per token
      if (d < minv) { minv = d; mini = code; }
    }
  }
  float ov = __shfl_xor(minv, 16);
  int oi = __shfl_xor(mini, 16);
  if (ov < minv || (ov == minv && oi < mini)) { minv = ov; mini = oi; }
  if (lane < 16) codes[(size_t)(tok0 + lane) * NCB_ + stage] = mini;

  // residual update + vq-loss partial; stage 5 also writes z_q_total
  const float* cbs32 = cb + (size_t)stage * KCB_ * DIM_;
  const bool last = (stage == NCB_ - 1);
  float lsum = 0.0f;
  for (int t = 0; t < 16; ++t) {
    int ci = __shfl(mini, t);
    const float* cv = cbs32 + (size_t)ci * DIM_;
    size_t base = (size_t)(tok0 + t) * DIM_;
#pragma unroll
    for (int i = 0; i < 3; ++i) {
      int d = lane + 32 * i;
      float q = cv[d];
      float nr = myR[t * DIM_ + d] - q;
      resid[base + d] = nr;
      lsum += nr * nr;
      if (last) zqt[base + d] = z[base + d] - nr;  // sum of all z_q stages
    }
  }
#pragma unroll
  for (int off = 16; off >= 1; off >>= 1) lsum += __shfl_down(lsum, off);
  if (lane == 0) atomicAdd(&acc[0], lsum);
}

// ---- stage-0 semantic MLP + MSE, fused per 16-token tile -------------------
// w2v target tile (48KB) is async-prefetched into LDS under GEMM1's WMMAs.
__global__ __launch_bounds__(128) void semantic_kernel(
    const float* __restrict__ z, const float* __restrict__ resid,
    const f16* __restrict__ w1t, const float* __restrict__ b1,
    const f16* __restrict__ w2t, const float* __restrict__ b2,
    const float* __restrict__ tgt, float* __restrict__ acc) {
  __shared__ alignas(16) float sTgt[16 * W2V_];
  __shared__ alignas(32) f16 sZ[16 * DIM_];
  __shared__ alignas(32) f16 sH[16 * PROJ_];
  __shared__ float sRed[4];

  const int tid = threadIdx.x;
  const int wave = tid >> 5;
  const int lane = tid & 31;
  const int tok0 = blockIdx.x * 16;

  // kick off async target tile load first (hidden under GEMM1)
  const float* tsrc = tgt + (size_t)tok0 * W2V_;
  for (int c = tid; c < (16 * W2V_) / 4; c += 128)
    async_cp16(tsrc + 4 * c, sTgt + 4 * c);

  // z_q0 = z - residual_after_stage0
  const size_t zb = (size_t)tok0 * DIM_;
  for (int e = tid; e < 16 * DIM_; e += 128)
    sZ[e] = (f16)(z[zb + e] - resid[zb + e]);
  __syncthreads();

  const int mrow = lane & 15;
  const int rhalf = (lane < 16) ? 0 : 8;

  // GEMM1 (16x192, K=96) + bias + exact gelu -> f16 h tile in LDS
  for (int nt = wave * 3; nt < wave * 3 + 3; ++nt) {
    v8f hacc = {};
#pragma unroll
    for (int ks = 0; ks < 3; ++ks) {
      v16h a = load_a16(sZ + mrow * DIM_ + ks * 32, lane);
      v16h b = load_b16(w1t + (size_t)(nt * 16 + mrow) * DIM_ + ks * 32, lane);
      hacc = __builtin_amdgcn_wmma_f32_16x16x32_f16(false, a, false, b,
                                                    (short)0, hacc, false, false);
    }
    int n = nt * 16 + mrow;
    float bias = b1[n];
#pragma unroll
    for (int v = 0; v < 8; ++v) {
      float x = hacc[v] + bias;
      float g = 0.5f * x * (1.0f + erff(x * 0.70710678f));
      sH[(v + rhalf) * PROJ_ + n] = (f16)g;
    }
  }
  wait_async();        // targets now resident in LDS
  __syncthreads();

  // GEMM2 (16x768, K=192) + bias, squared error vs LDS targets
  float lsum = 0.0f;
  for (int nt = wave * 12; nt < wave * 12 + 12; ++nt) {
    v8f pacc = {};
#pragma unroll
    for (int ks = 0; ks < 6; ++ks) {
      v16h a = load_a16(sH + mrow * PROJ_ + ks * 32, lane);
      v16h b = load_b16(w2t + (size_t)(nt * 16 + mrow) * PROJ_ + ks * 32, lane);
      pacc = __builtin_amdgcn_wmma_f32_16x16x32_f16(false, a, false, b,
                                                    (short)0, pacc, false, false);
    }
    int n = nt * 16 + mrow;
    float bias = b2[n];
#pragma unroll
    for (int v = 0; v < 8; ++v) {
      float e = (pacc[v] + bias) - sTgt[(v + rhalf) * W2V_ + n];
      lsum += e * e;
    }
  }
#pragma unroll
  for (int off = 16; off >= 1; off >>= 1) lsum += __shfl_down(lsum, off);
  if (lane == 0) sRed[wave] = lsum;
  __syncthreads();
  if (tid == 0) atomicAdd(&acc[1], sRed[0] + sRed[1] + sRed[2] + sRed[3]);
}

__global__ void finalize_kernel(const float* __restrict__ acc, float* __restrict__ outs) {
  if (threadIdx.x == 0 && blockIdx.x == 0) {
    outs[0] = 1.5f * acc[0] / (float)((size_t)BT_ * DIM_);  // vq_loss
    outs[1] = acc[1] / (float)((size_t)BT_ * W2V_);         // semantic_loss
  }
}

// ---- host driver -----------------------------------------------------------
extern "C" void kernel_launch(void* const* d_in, const int* in_sizes, int n_in,
                              void* d_out, int out_size, void* d_ws, size_t ws_size,
                              hipStream_t stream) {
  const float* z   = (const float*)d_in[0];
  const float* w2v = (const float*)d_in[1];
  const float* cb  = (const float*)d_in[2];
  const float* w1  = (const float*)d_in[3];
  const float* b1  = (const float*)d_in[4];
  const float* w2  = (const float*)d_in[5];
  const float* b2  = (const float*)d_in[6];

  float* out = (float*)d_out;
  float* zqt = out;                                             // (B,T,D)
  int* codes = (int*)(out + (size_t)BT_ * DIM_);                // (B,T,NCB)
  float* scal = out + (size_t)BT_ * DIM_ + (size_t)BT_ * NCB_;  // losses

  char* ws = (char*)d_ws;
  float* resid = (float*)(ws + WS_RESID);
  f16* cbh     = (f16*)(ws + WS_CBH);
  f16* w1t     = (f16*)(ws + WS_W1T);
  f16* w2t     = (f16*)(ws + WS_W2T);
  float* norms = (float*)(ws + WS_NORM);
  float* acc   = (float*)(ws + WS_ACC);

  init_kernel<<<2048, 256, 0, stream>>>(z, resid, acc);
  cvt_cb_kernel<<<(NCB_ * KCB_ * DIM_ + 255) / 256, 256, 0, stream>>>(cb, cbh);
  norms_kernel<<<(NCB_ * KCB_ + 255) / 256, 256, 0, stream>>>(cb, norms);
  cvt_w_kernel<<<(PROJ_ * DIM_ + W2V_ * PROJ_ + 255) / 256, 256, 0, stream>>>(w1, w2, w1t, w2t);

  for (int s = 0; s < NCB_; ++s) {
    vq_stage_kernel<<<BT_ / 128, 256, 0, stream>>>(cb, cbh, norms, z, resid, zqt,
                                                   codes, acc, s);
    if (s == 0)  // resid now holds residual after stage 0
      semantic_kernel<<<BT_ / 16, 128, 0, stream>>>(z, resid, w1t, b1, w2t, b2, w2v, acc);
  }
  finalize_kernel<<<1, 1, 0, stream>>>(acc, scal);
}
